// TriangleMultiplication_15616501088403
// MI455X (gfx1250) — matrix operand order repaired and checked by reading, not compiled, not run on hardware
//
#include <hip/hip_runtime.h>
#include <hip/hip_bf16.h>

// ---------------------------------------------------------------------------
// Triangle multiplication (outgoing) for MI455X / gfx1250.
// All contractions run on v_wmma_f32_16x16x32_bf16 (wave32), intermediates in
// bf16 to halve HBM traffic (workload is bandwidth-bound: ~1.2GB moved,
// ~172 GFLOP compute). LDS staging for A-fragments; all WMMA fragment loads
// are contiguous 16-byte vectors per the CDNA5 VGPR layout tables.
// The triangle kernel stages its A-tile with the Tensor Data Mover
// (tensor_load_to_lds + s_wait_tensorcnt), falling back to cooperative
// copies if the builtin is unavailable.
// ---------------------------------------------------------------------------

#define NDIM 512
#define DDIM 256
#define CDIM 128
#define NN   (512LL * 512LL)

typedef __attribute__((ext_vector_type(16))) __bf16 v16bf;
typedef __attribute__((ext_vector_type(8)))  __bf16 v8bf;
typedef __attribute__((ext_vector_type(4)))  __bf16 v4bf;
typedef __attribute__((ext_vector_type(8)))  float  v8f;
typedef __attribute__((ext_vector_type(4)))  unsigned int v4u;
typedef __attribute__((ext_vector_type(8)))  int    v8i;
typedef __attribute__((ext_vector_type(4)))  int    v4i;

union ABfrag { v16bf v; v8bf h[2]; };

static __device__ inline v8f zero8() {
    v8f z;
#pragma unroll
    for (int i = 0; i < 8; ++i) z[i] = 0.0f;
    return z;
}

static __device__ inline float sigmoidf(float v) {
    return 1.0f / (1.0f + __expf(-v));
}

static __device__ inline float wave_sum(float v) {
#pragma unroll
    for (int o = 16; o > 0; o >>= 1) v += __shfl_xor(v, o, 32);
    return v;
}

static __device__ inline v8f wmma_bf16(const ABfrag& a, const ABfrag& b, v8f c) {
    return __builtin_amdgcn_wmma_f32_16x16x32_bf16(
        /*neg_a=*/false, a.v, /*neg_b=*/false, b.v,
        /*c_mod=*/(short)0, c, /*reuse_a=*/false, /*reuse_b=*/false);
}

// ---------------------------------------------------------------------------
// K0: f32 -> bf16 weight conversion (weights then live in the 192MB L2)
// ---------------------------------------------------------------------------
__global__ void cvt_kernel(const float* __restrict__ src, __bf16* __restrict__ dst, int n) {
    int i = blockIdx.x * blockDim.x + threadIdx.x;
    if (i < n) dst[i] = (__bf16)src[i];
}

// ---------------------------------------------------------------------------
// K1: masked, gated left/right projections.
//   l[c][p] = m2 * (x[p,:]·Wl[c,:]) * sigmoid(x[p,:]·Wlg[c,:])   (bf16 out)
// One block = 16 consecutive pairs p (same i row), 8 waves x 16 channels.
// ---------------------------------------------------------------------------
__global__ __launch_bounds__(256) void proj_kernel(
    const float* __restrict__ x, const int* __restrict__ mask,
    const __bf16* __restrict__ wl,  const __bf16* __restrict__ wr,
    const __bf16* __restrict__ wlg, const __bf16* __restrict__ wrg,
    __bf16* __restrict__ lw, __bf16* __restrict__ rw)
{
    __shared__ __bf16 sx[16 * DDIM];   // 8 KB, x slab as bf16
    __shared__ float  sm[17];          // 16 j-masks + 1 i-mask

    const int p0 = blockIdx.x * 16;
    const int t  = threadIdx.x;

    // stage x[p0..p0+15, 0..255] -> bf16 LDS (thread t: 16 contiguous floats)
    {
        const float4* src = reinterpret_cast<const float4*>(x + (size_t)p0 * DDIM) + t * 4;
        v8bf* dst = reinterpret_cast<v8bf*>(&sx[t * 16]);
#pragma unroll
        for (int q = 0; q < 2; ++q) {
            float4 a = src[q * 2 + 0];
            float4 b = src[q * 2 + 1];
            v8bf pk;
            pk[0] = (__bf16)a.x; pk[1] = (__bf16)a.y; pk[2] = (__bf16)a.z; pk[3] = (__bf16)a.w;
            pk[4] = (__bf16)b.x; pk[5] = (__bf16)b.y; pk[6] = (__bf16)b.z; pk[7] = (__bf16)b.w;
            dst[q] = pk;
        }
    }
    if (t < 16) sm[t] = 1.0f - (float)mask[(p0 & (NDIM - 1)) + t];
    if (t == 16) sm[16] = 1.0f - (float)mask[p0 >> 9];
    __syncthreads();

    const int lane = t & 31, w = t >> 5;
    const int n = lane & 15, half = lane >> 4;
    const int c = w * 16 + n;

    v8f accl = zero8(), acclg = zero8(), accr = zero8(), accrg = zero8();
    const __bf16* wl_row  = wl  + c * DDIM;
    const __bf16* wr_row  = wr  + c * DDIM;
    const __bf16* wlg_row = wlg + c * DDIM;
    const __bf16* wrg_row = wrg + c * DDIM;

#pragma unroll
    for (int ks = 0; ks < 8; ++ks) {
        ABfrag a, b;
        const int ka = ks * 32 + half * 16;           // A: one 16-elem run
        a.h[0] = *reinterpret_cast<const v8bf*>(&sx[n * DDIM + ka]);
        a.h[1] = *reinterpret_cast<const v8bf*>(&sx[n * DDIM + ka + 8]);
        const int kb = ks * 32 + half * 16;           // B: one 16-elem run

        b.h[0] = *reinterpret_cast<const v8bf*>(wl_row + kb);
        b.h[1] = *reinterpret_cast<const v8bf*>(wl_row + kb + 8);
        accl = wmma_bf16(a, b, accl);

        b.h[0] = *reinterpret_cast<const v8bf*>(wlg_row + kb);
        b.h[1] = *reinterpret_cast<const v8bf*>(wlg_row + kb + 8);
        acclg = wmma_bf16(a, b, acclg);

        b.h[0] = *reinterpret_cast<const v8bf*>(wr_row + kb);
        b.h[1] = *reinterpret_cast<const v8bf*>(wr_row + kb + 8);
        accr = wmma_bf16(a, b, accr);

        b.h[0] = *reinterpret_cast<const v8bf*>(wrg_row + kb);
        b.h[1] = *reinterpret_cast<const v8bf*>(wrg_row + kb + 8);
        accrg = wmma_bf16(a, b, accrg);
    }

    const float mi = sm[16];
    v8bf lp, rp;
#pragma unroll
    for (int v = 0; v < 8; ++v) {
        const int m = v + half * 8;                   // D-matrix row
        const float s = mi * sm[m];
        lp[v] = (__bf16)(s * accl[v] * sigmoidf(acclg[v]));
        rp[v] = (__bf16)(s * accr[v] * sigmoidf(accrg[v]));
    }
    // channel-major [C][N][N]: 8 consecutive p per lane -> one 16B store each
    const size_t pofs = (size_t)c * NN + (size_t)(p0 + half * 8);
    *reinterpret_cast<v8bf*>(lw + pofs) = lp;
    *reinterpret_cast<v8bf*>(rw + pofs) = rp;
}

// ---------------------------------------------------------------------------
// K2: triangle contraction o[i,j,c] = sum_k l[c][i,k] * r[c][j,k].
// grid = (c=128, i_block=16, j_chunk=4); block stages a 32x512 l-tile via the
// Tensor Data Mover; each wave computes two 16x16 (i,j) tiles (32 WMMAs),
// reusing every B fragment twice.
// ---------------------------------------------------------------------------
__global__ __launch_bounds__(256) void tri_kernel(
    const __bf16* __restrict__ lw, const __bf16* __restrict__ rw,
    __bf16* __restrict__ ow)
{
    __shared__ __bf16 sl[32 * NDIM];   // 32 KB: l[c][i0..i0+31][0..511]

    const int c  = blockIdx.x;
    const int i0 = blockIdx.y * 32;
    const int t  = threadIdx.x;
    const size_t plane = (size_t)c * NN;
    const __bf16* ltile = lw + plane + (size_t)i0 * NDIM;

#if __has_builtin(__builtin_amdgcn_tensor_load_to_lds)
    if (t < 32) {   // wave 0 issues one TDM descriptor for the whole tile
        const unsigned long long ga = (unsigned long long)(size_t)ltile;
        const unsigned int lds_base = (unsigned int)(size_t)(&sl[0]);

        v4u g0;                                   // D# group 0 (128b)
        g0[0] = 1u;                               // count=1, user descriptor
        g0[1] = lds_base;                         // lds_addr
        g0[2] = (unsigned int)ga;                 // global_addr[31:0]
        g0[3] = (unsigned int)((ga >> 32) & 0x1FFFFFFu) | (2u << 30); // type=2

        v8i g1;                                   // D# group 1 (256b)
        g1[0] = 1 << 16;                          // data_size=1 (2 bytes)
        g1[1] = (int)(512u << 16);                // tensor_dim0 = 512 (lo16)
        g1[2] = (int)(512u << 16);                // tensor_dim1 = 512 (lo16)
        g1[3] = (int)(512u << 16);                // tile_dim0 = 512
        g1[4] = 32;                               // tile_dim1 = 32, tile_dim2 = 0
        g1[5] = 512;                              // tensor_dim0_stride = 512
        g1[6] = 0;
        g1[7] = 0;

        v4i g2 = {0, 0, 0, 0};
        v4i g3 = {0, 0, 0, 0};
#if defined(__clang_major__) && (__clang_major__ >= 23)
        v8i g4 = {0, 0, 0, 0, 0, 0, 0, 0};
        __builtin_amdgcn_tensor_load_to_lds(g0, g1, g2, g3, g4, 0);
#else
        __builtin_amdgcn_tensor_load_to_lds(g0, g1, g2, g3, 0);
#endif
        __builtin_amdgcn_s_wait_tensorcnt(0);
    }
    __syncthreads();
#else
    {   // cooperative fallback: thread t copies 64 contiguous bf16 (128B)
        const int base = t * 64;
        const int row = base >> 9, col = base & (NDIM - 1);
        const v8bf* src = reinterpret_cast<const v8bf*>(ltile + (size_t)row * NDIM + col);
        v8bf* dst = reinterpret_cast<v8bf*>(&sl[base]);
#pragma unroll
        for (int q = 0; q < 8; ++q) dst[q] = src[q];
    }
    __syncthreads();
#endif

    const int lane = t & 31, w = t >> 5;
    const int n = lane & 15, half = lane >> 4;
    const int j = (blockIdx.z * 8 + w) * 16 + n;
    const __bf16* r_row = rw + plane + (size_t)j * NDIM;

    __builtin_prefetch(r_row, 0, 0);   // global_prefetch_b8 into L2

    v8f acc0 = zero8(), acc1 = zero8();
#pragma unroll
    for (int ks = 0; ks < 16; ++ks) {
        ABfrag a0, a1, b;
        const int kb = ks * 32 + half * 8;
        b.h[0] = *reinterpret_cast<const v8bf*>(r_row + kb);
        b.h[1] = *reinterpret_cast<const v8bf*>(r_row + kb + 16);
        const int ka = ks * 32 + half * 8;
        a0.h[0] = *reinterpret_cast<const v8bf*>(&sl[n * NDIM + ka]);
        a0.h[1] = *reinterpret_cast<const v8bf*>(&sl[n * NDIM + ka + 16]);
        a1.h[0] = *reinterpret_cast<const v8bf*>(&sl[(n + 16) * NDIM + ka]);
        a1.h[1] = *reinterpret_cast<const v8bf*>(&sl[(n + 16) * NDIM + ka + 16]);
        acc0 = wmma_bf16(a0, b, acc0);
        acc1 = wmma_bf16(a1, b, acc1);
    }

#pragma unroll
    for (int v = 0; v < 8; ++v) {
        const int ia = i0 + v + half * 8;
        const int ib = ia + 16;
        ow[((size_t)ia * NDIM + j) * CDIM + c] = (__bf16)acc0[v];   // [p][C]
        ow[((size_t)ib * NDIM + j) * CDIM + c] = (__bf16)acc1[v];
    }
}

// ---------------------------------------------------------------------------
// K3: LayerNorm over C=128, in place, one wave per pair (wave32 shfl reduce).
// ---------------------------------------------------------------------------
__global__ __launch_bounds__(256) void ln_kernel(
    __bf16* __restrict__ ow,
    const float* __restrict__ gamma, const float* __restrict__ beta)
{
    const int t = threadIdx.x, lane = t & 31, w = t >> 5;
    const size_t p = (size_t)blockIdx.x * 8 + w;
    __bf16* rowp = ow + p * CDIM + lane * 4;

    v4bf d = *reinterpret_cast<const v4bf*>(rowp);
    float f[4];
#pragma unroll
    for (int q = 0; q < 4; ++q) f[q] = (float)d[q];

    const float s  = wave_sum(f[0] + f[1] + f[2] + f[3]);
    const float s2 = wave_sum(f[0]*f[0] + f[1]*f[1] + f[2]*f[2] + f[3]*f[3]);
    const float mu   = s * (1.0f / CDIM);
    const float var  = s2 * (1.0f / CDIM) - mu * mu;
    const float rstd = rsqrtf(var + 1e-5f);

    const int cb = lane * 4;
    v4bf o;
#pragma unroll
    for (int q = 0; q < 4; ++q)
        o[q] = (__bf16)((f[q] - mu) * rstd * gamma[cb + q] + beta[cb + q]);
    *reinterpret_cast<v4bf*>(rowp) = o;
}

// ---------------------------------------------------------------------------
// K4: out[p,d] = (o_ln[p,:]·Wo[d,:]) * sigmoid(x[p,:]·Wog[d,:]), f32 out.
// One block = 16 pairs; 8 waves x 2 d-tiles of 16 each (covers D=256).
// ---------------------------------------------------------------------------
__global__ __launch_bounds__(256) void out_kernel(
    const float* __restrict__ x, const __bf16* __restrict__ oln,
    const __bf16* __restrict__ wo, const __bf16* __restrict__ wog,
    float* __restrict__ out)
{
    __shared__ __bf16 sx[16 * DDIM];   // 8 KB
    __shared__ __bf16 so[16 * CDIM];   // 4 KB

    const int p0 = blockIdx.x * 16;
    const int t  = threadIdx.x;

    {   // stage x slab as bf16
        const float4* src = reinterpret_cast<const float4*>(x + (size_t)p0 * DDIM) + t * 4;
        v8bf* dst = reinterpret_cast<v8bf*>(&sx[t * 16]);
#pragma unroll
        for (int q = 0; q < 2; ++q) {
            float4 a = src[q * 2 + 0];
            float4 b = src[q * 2 + 1];
            v8bf pk;
            pk[0] = (__bf16)a.x; pk[1] = (__bf16)a.y; pk[2] = (__bf16)a.z; pk[3] = (__bf16)a.w;
            pk[4] = (__bf16)b.x; pk[5] = (__bf16)b.y; pk[6] = (__bf16)b.z; pk[7] = (__bf16)b.w;
            dst[q] = pk;
        }
    }
    {   // stage o_ln slab (already bf16): thread t copies 16B
        reinterpret_cast<v8bf*>(so)[t] =
            reinterpret_cast<const v8bf*>(oln + (size_t)p0 * CDIM)[t];
    }
    __syncthreads();

    const int lane = t & 31, w = t >> 5;
    const int n = lane & 15, half = lane >> 4;

#pragma unroll
    for (int dt = 0; dt < 2; ++dt) {
        const int d = (w * 2 + dt) * 16 + n;
        const __bf16* wo_row  = wo  + d * CDIM;
        const __bf16* wog_row = wog + d * DDIM;
        v8f ao = zero8(), ag = zero8();

#pragma unroll
        for (int ks = 0; ks < 4; ++ks) {      // output projection, K = 128
            ABfrag a, b;
            const int ka = ks * 32 + half * 16;
            a.h[0] = *reinterpret_cast<const v8bf*>(&so[n * CDIM + ka]);
            a.h[1] = *reinterpret_cast<const v8bf*>(&so[n * CDIM + ka + 8]);
            const int kb = ks * 32 + half * 16;
            b.h[0] = *reinterpret_cast<const v8bf*>(wo_row + kb);
            b.h[1] = *reinterpret_cast<const v8bf*>(wo_row + kb + 8);
            ao = wmma_bf16(a, b, ao);
        }
#pragma unroll
        for (int ks = 0; ks < 8; ++ks) {      // output gate, K = 256
            ABfrag a, b;
            const int ka = ks * 32 + half * 16;
            a.h[0] = *reinterpret_cast<const v8bf*>(&sx[n * DDIM + ka]);
            a.h[1] = *reinterpret_cast<const v8bf*>(&sx[n * DDIM + ka + 8]);
            const int kb = ks * 32 + half * 16;
            b.h[0] = *reinterpret_cast<const v8bf*>(wog_row + kb);
            b.h[1] = *reinterpret_cast<const v8bf*>(wog_row + kb + 8);
            ag = wmma_bf16(a, b, ag);
        }

#pragma unroll
        for (int v = 0; v < 8; ++v) {
            const int p = p0 + v + half * 8;
            out[(size_t)p * DDIM + d] = ao[v] * sigmoidf(ag[v]);
        }
    }
}

// ---------------------------------------------------------------------------
// Host-side launcher. Workspace layout (bytes):
//   [0, 448K)        bf16 weights: Wl, Wr, Wlg, Wrg (64K each), Wo (64K), Wog (128K)
//   [512K, +64MB)    l  bf16 [C][N][N]
//   [.. , +64MB)     r  bf16 [C][N][N]
//   [.. , +64MB)     o  bf16 [N*N][C]   (LayerNorm applied in place)
// Total ~192.5 MB.
// ---------------------------------------------------------------------------
extern "C" void kernel_launch(void* const* d_in, const int* in_sizes, int n_in,
                              void* d_out, int out_size, void* d_ws, size_t ws_size,
                              hipStream_t stream) {
    const float* x    = (const float*)d_in[0];
    const int*   mask = (const int*)  d_in[1];
    const float* Wl   = (const float*)d_in[2];
    const float* Wr   = (const float*)d_in[3];
    const float* Wlg  = (const float*)d_in[4];
    const float* Wrg  = (const float*)d_in[5];
    const float* gam  = (const float*)d_in[6];
    const float* bet  = (const float*)d_in[7];
    const float* Wo   = (const float*)d_in[8];
    const float* Wog  = (const float*)d_in[9];
    float* out = (float*)d_out;

    char* ws = (char*)d_ws;
    __bf16* wl_b  = (__bf16*)(ws + 0);
    __bf16* wr_b  = (__bf16*)(ws + 65536);
    __bf16* wlg_b = (__bf16*)(ws + 131072);
    __bf16* wrg_b = (__bf16*)(ws + 196608);
    __bf16* wo_b  = (__bf16*)(ws + 262144);
    __bf16* wog_b = (__bf16*)(ws + 327680);

    const size_t PLANE_B = (size_t)CDIM * NN * 2;   // 64 MB
    __bf16* l_b = (__bf16*)(ws + (1u << 19));
    __bf16* r_b = (__bf16*)(ws + (1u << 19) + PLANE_B);
    __bf16* o_b = (__bf16*)(ws + (1u << 19) + 2 * PLANE_B);

    const int WCD = CDIM * DDIM;    // 32768
    cvt_kernel<<<(WCD + 255) / 256, 256, 0, stream>>>(Wl,  wl_b,  WCD);
    cvt_kernel<<<(WCD + 255) / 256, 256, 0, stream>>>(Wr,  wr_b,  WCD);
    cvt_kernel<<<(WCD + 255) / 256, 256, 0, stream>>>(Wlg, wlg_b, WCD);
    cvt_kernel<<<(WCD + 255) / 256, 256, 0, stream>>>(Wrg, wrg_b, WCD);
    cvt_kernel<<<(WCD + 255) / 256, 256, 0, stream>>>(Wo,  wo_b,  WCD);
    cvt_kernel<<<(DDIM * DDIM + 255) / 256, 256, 0, stream>>>(Wog, wog_b, DDIM * DDIM);

    proj_kernel<<<(int)(NN / 16), 256, 0, stream>>>(x, mask, wl_b, wr_b, wlg_b, wrg_b, l_b, r_b);
    tri_kernel<<<dim3(CDIM, NDIM / 32, 4), 256, 0, stream>>>(l_b, r_b, o_b);
    ln_kernel<<<(int)(NN / 8), 256, 0, stream>>>(o_b, gam, bet);
    out_kernel<<<(int)(NN / 16), 256, 0, stream>>>(x, o_b, wo_b, wog_b, out);
}
